// LDR_TGN_78984448573534
// MI455X (gfx1250) — matrix operand order repaired and checked by reading, not compiled, not run on hardware
//
#include <hip/hip_runtime.h>
#include <math.h>

#define NTOK  200000
#define NPOOL 50000
#define NEDGE 500000
#define HID   64
#define HEADS 4
#define TOKF  4
#define POOLF 8

typedef __attribute__((ext_vector_type(16))) __bf16 v16bf;
typedef __attribute__((ext_vector_type(8)))  __bf16 v8bf;
typedef __attribute__((ext_vector_type(8)))  float  v8f;

// ---------------- helpers ----------------

__device__ __forceinline__ __bf16 f2bf(float f) {
  unsigned u = __float_as_uint(f);
  unsigned r = u + 0x7FFFu + ((u >> 16) & 1u);   // round-to-nearest-even
  unsigned short s = (unsigned short)(r >> 16);
  __bf16 b; __builtin_memcpy(&b, &s, 2); return b;
}

// order-preserving float <-> uint for atomicMax-based segment max
__device__ __forceinline__ unsigned fenc(float f) {
  unsigned u = __float_as_uint(f);
  return (u & 0x80000000u) ? ~u : (u | 0x80000000u);
}
__device__ __forceinline__ float fdec(unsigned u) {
  return (u & 0x80000000u) ? __uint_as_float(u ^ 0x80000000u)
                           : __uint_as_float(~u);
}

__device__ __forceinline__ float sigmoidf_(float x) { return 1.f / (1.f + expf(-x)); }

__device__ __forceinline__ float4 ldf4(const float* p) {
  return *reinterpret_cast<const float4*>(p);
}

// ---------------- WMMA GEMM ----------------
// C[n,m] = sum_k A[n,k]*Wb[k,m] + bias[m], A and Wb bf16, f32 accumulate.
// Preconditions: N%16==0; K (template) = zero-padded K of A (lda>=K) and Wb;
// Mp = ldw multiple of 16; bias has Mp entries; all Mp columns stored (ldc>=Mp).
// One wave = one 16x16 tile. Inner loop: 2x16B A loads + 16x2B strided B loads + 1 WMMA.

template <int K, bool BFOUT>
__global__ __launch_bounds__(256) void wmma_gemm_bias(
    const __bf16* __restrict__ A, int lda,
    const __bf16* __restrict__ Wb, int ldw,
    const float* __restrict__ bias,
    void* __restrict__ Cout, int ldc, int N)
{
  const int lane = threadIdx.x & 31;
  const int wave = threadIdx.x >> 5;
  const int mt = ldw >> 4;
  const int tile = blockIdx.x * 8 + wave;
  if (tile >= (N >> 4) * mt) return;
  const int tr = tile / mt, tc = tile % mt;
  const int row0 = tr << 4, col0 = tc << 4;

  const int mloc = lane & 15;
  const int hi   = lane >> 4;                 // K-half select (ISA A layout)
  const __bf16* __restrict__ arow = A + (size_t)(row0 + mloc) * lda;

  v8f acc = {0.f, 0.f, 0.f, 0.f, 0.f, 0.f, 0.f, 0.f};

  #pragma unroll
  for (int k0 = 0; k0 < K; k0 += 32) {
    // A fragment: lane half 'hi' holds K-chunks [k0+8hi, +8) and [k0+16+8hi, +8)
    v8bf alo = *reinterpret_cast<const v8bf*>(arow + k0 + (hi << 3));
    v8bf ahi = *reinterpret_cast<const v8bf*>(arow + k0 + 16 + (hi << 3));
    v16bf a = __builtin_shufflevector(alo, ahi,
        0, 1, 2, 3, 4, 5, 6, 7, 8, 9, 10, 11, 12, 13, 14, 15);
    // B fragment: lane l holds column col0+(l&15), K in same order as A (k0+8hi+s, k0+16+8hi+s-8)
    v16bf b;
    #pragma unroll
    for (int s = 0; s < 16; ++s)
      b[s] = Wb[(size_t)(k0 + (s < 8 ? (hi << 3) + s : 16 + (hi << 3) + (s - 8))) * ldw + col0 + mloc];
    acc = __builtin_amdgcn_wmma_f32_16x16x32_bf16(
        false, a, false, b, (short)0, acc, false, false);
  }

  const int ccol = col0 + mloc;
  const float bs = bias[ccol];
  #pragma unroll
  for (int r = 0; r < 8; ++r) {
    const int rr = row0 + r + (hi << 3);
    if (BFOUT) ((__bf16*)Cout)[(size_t)rr * ldc + ccol] = f2bf(acc[r] + bs);
    else       ((float*)Cout)[(size_t)rr * ldc + ccol] = acc[r] + bs;
  }
}

// ---------------- weight / bias packing (f32 -> padded bf16 / padded f32) ----------------

__global__ void pack_wbf(const float* __restrict__ W, int K, int M,
                         __bf16* __restrict__ out, int Kp, int Mp)
{
  long long t = (long long)blockIdx.x * blockDim.x + threadIdx.x;
  if (t >= (long long)Kp * Mp) return;
  int k = (int)(t / Mp), m = (int)(t % Mp);
  float v = (k < K && m < M) ? W[(size_t)k * M + m] : 0.f;
  out[t] = f2bf(v);
}

__global__ void pack_bias(const float* __restrict__ b, int M,
                          float* __restrict__ out, int Mp)
{
  int t = blockIdx.x * blockDim.x + threadIdx.x;
  if (t < Mp) out[t] = (t < M) ? b[t] : 0.f;
}

// ---------------- LayerNorm(+prebias) + ELU over 64-dim rows -> bf16; wave per row ----------------

__global__ __launch_bounds__(256) void ln_elu_kernel(
    const float* __restrict__ X, const float* __restrict__ prebias,
    const float* __restrict__ g, const float* __restrict__ b,
    __bf16* __restrict__ Y, int N)
{
  const int lane = threadIdx.x & 31;
  const int row  = blockIdx.x * 8 + (threadIdx.x >> 5);
  if (row >= N) return;
  const float* x = X + (size_t)row * 64;
  float v0 = x[lane]      + (prebias ? prebias[lane]      : 0.f);
  float v1 = x[lane + 32] + (prebias ? prebias[lane + 32] : 0.f);
  float s = v0 + v1;
  for (int m = 16; m >= 1; m >>= 1) s += __shfl_xor(s, m, 32);
  float mu = s * (1.f / 64.f);
  float d0 = v0 - mu, d1 = v1 - mu;
  float q = d0 * d0 + d1 * d1;
  for (int m = 16; m >= 1; m >>= 1) q += __shfl_xor(q, m, 32);
  float rs = rsqrtf(q * (1.f / 64.f) + 1e-5f);
  float y0 = d0 * rs * g[lane]      + b[lane];
  float y1 = d1 * rs * g[lane + 32] + b[lane + 32];
  y0 = (y0 > 0.f) ? y0 : (expf(y0) - 1.f);
  y1 = (y1 > 0.f) ? y1 : (expf(y1) - 1.f);
  __bf16* y = Y + (size_t)row * 64;
  y[lane] = f2bf(y0); y[lane + 32] = f2bf(y1);
}

// ---------------- misc elementwise ----------------

__global__ void fillf(float* __restrict__ p, float v, long long n) {
  long long t = (long long)blockIdx.x * blockDim.x + threadIdx.x;
  if (t < n) p[t] = v;
}

// xinb[n, 0:F)=feats, [F,F+64)=memtab[ids[n]], [F+64,96)=0 (bf16);
// also memf/memb = gathered memory in f32/bf16 (stride 64)
__global__ void concat_gather(const float* __restrict__ feats, int F,
                              const int* __restrict__ ids,
                              const float* __restrict__ memtab,
                              __bf16* __restrict__ xinb,
                              float* __restrict__ memf,
                              __bf16* __restrict__ memb, int N)
{
  long long t = (long long)blockIdx.x * blockDim.x + threadIdx.x;
  if (t >= (long long)N * 96) return;
  int n = (int)(t / 96), j = (int)(t % 96);
  float v = 0.f;
  if (j < F) v = feats[(size_t)n * F + j];
  else if (j < F + 64) v = memtab[(size_t)ids[n] * 64 + (j - F)];
  xinb[t] = f2bf(v);
  if (j >= F && j < F + 64) {
    memf[(size_t)n * 64 + (j - F)] = v;
    memb[(size_t)n * 64 + (j - F)] = f2bf(v);
  }
}

// out[n, 0:FA)=A, [FA, FA+64)=B, [FA+64, 96)=0  (bf16)
__global__ void concat2_bf(const float* __restrict__ A, int FA,
                           const float* __restrict__ B,
                           __bf16* __restrict__ out, int N)
{
  long long t = (long long)blockIdx.x * blockDim.x + threadIdx.x;
  if (t >= (long long)N * 96) return;
  int n = (int)(t / 96), j = (int)(t % 96);
  float v = 0.f;
  if (j < FA) v = A[(size_t)n * FA + j];
  else if (j < FA + 64) v = B[(size_t)n * 64 + (j - FA)];
  out[t] = f2bf(v);
}

// ---------------- GATv2 edge passes (thread per edge*head) ----------------

template <int C>
__global__ void edge_logits_max(const float* __restrict__ xl,
                                const float* __restrict__ xr,
                                const float* __restrict__ att,
                                const int* __restrict__ src,
                                const int* __restrict__ dst,
                                float* __restrict__ logits,
                                unsigned* __restrict__ segm, int E)
{
  int t = blockIdx.x * blockDim.x + threadIdx.x;
  if (t >= E * HEADS) return;
  int e = t >> 2, h = t & 3;
  int s = src[e], d = dst[e];
  const float* pl = xl + (size_t)s * (HEADS * C) + h * C;
  const float* pr = xr + (size_t)d * (HEADS * C) + h * C;
  const float* pa = att + h * C;
  float acc = 0.f;
  #pragma unroll
  for (int c = 0; c < C; c += 4) {
    float4 l = ldf4(pl + c), r = ldf4(pr + c), a = ldf4(pa + c);
    float v;
    v = l.x + r.x; v = (v > 0.f) ? v : 0.2f * v; acc += a.x * v;
    v = l.y + r.y; v = (v > 0.f) ? v : 0.2f * v; acc += a.y * v;
    v = l.z + r.z; v = (v > 0.f) ? v : 0.2f * v; acc += a.z * v;
    v = l.w + r.w; v = (v > 0.f) ? v : 0.2f * v; acc += a.w * v;
  }
  logits[t] = acc;
  atomicMax(&segm[(size_t)d * HEADS + h], fenc(acc));
}

__global__ void edge_exp_sum(const int* __restrict__ dst,
                             float* __restrict__ logits,
                             const unsigned* __restrict__ segm,
                             float* __restrict__ segd, int E)
{
  int t = blockIdx.x * blockDim.x + threadIdx.x;
  if (t >= E * HEADS) return;
  int e = t >> 2, h = t & 3;
  int d = dst[e];
  float m = fdec(segm[(size_t)d * HEADS + h]);
  float ex = expf(logits[t] - m);
  logits[t] = ex;
  atomicAdd(&segd[(size_t)d * HEADS + h], ex);
}

template <int C>
__global__ void edge_scatter(const float* __restrict__ xl,
                             const int* __restrict__ src,
                             const int* __restrict__ dst,
                             const float* __restrict__ logits,
                             const float* __restrict__ segd,
                             float* __restrict__ outraw, int E)
{
  int t = blockIdx.x * blockDim.x + threadIdx.x;
  if (t >= E * HEADS) return;
  int e = t >> 2, h = t & 3;
  int s = src[e], d = dst[e];
  float w = logits[t] / (segd[(size_t)d * HEADS + h] + 1e-16f);
  const float* pl = xl + (size_t)s * (HEADS * C) + h * C;
  float* po = outraw + (size_t)d * (HEADS * C) + h * C;
  #pragma unroll
  for (int c = 0; c < C; c += 4) {
    float4 l = ldf4(pl + c);
    atomicAdd(po + c + 0, w * l.x);
    atomicAdd(po + c + 1, w * l.y);
    atomicAdd(po + c + 2, w * l.z);
    atomicAdd(po + c + 3, w * l.w);
  }
}

// conv2 epilogue: mean over 4 heads + bias -> d_out region (float4 per thread)
__global__ void head_mean_bias(const float* __restrict__ raw,
                               const float* __restrict__ bias,
                               float* __restrict__ out, int N)
{
  long long t = (long long)blockIdx.x * blockDim.x + threadIdx.x;
  if (t >= (long long)N * 16) return;
  int n = (int)(t >> 4), c = ((int)(t & 15)) << 2;
  const float* r = raw + (size_t)n * 256;
  float4 a = ldf4(r + c), b = ldf4(r + 64 + c), cc = ldf4(r + 128 + c), dd = ldf4(r + 192 + c);
  float4 bi = ldf4(bias + c);
  float4 o;
  o.x = 0.25f * (a.x + b.x + cc.x + dd.x) + bi.x;
  o.y = 0.25f * (a.y + b.y + cc.y + dd.y) + bi.y;
  o.z = 0.25f * (a.z + b.z + cc.z + dd.z) + bi.z;
  o.w = 0.25f * (a.w + b.w + cc.w + dd.w) + bi.w;
  *reinterpret_cast<float4*>(out + (size_t)n * 64 + c) = o;
}

// GRU cell elementwise, 4 channels per thread; h is f32 (stride 64)
__global__ void gru_kernel(const float* __restrict__ gi,
                           const float* __restrict__ gh,
                           const float* __restrict__ h,
                           float* __restrict__ out, int N)
{
  long long t = (long long)blockIdx.x * blockDim.x + threadIdx.x;
  if (t >= (long long)N * 16) return;
  int n = (int)(t >> 4), j = ((int)(t & 15)) << 2;
  const float* pi = gi + (size_t)n * 192;
  const float* ph = gh + (size_t)n * 192;
  float4 ir = ldf4(pi + j),        hr = ldf4(ph + j);
  float4 iz = ldf4(pi + 64 + j),   hz = ldf4(ph + 64 + j);
  float4 in = ldf4(pi + 128 + j),  hn = ldf4(ph + 128 + j);
  float4 hv = ldf4(h + (size_t)n * 64 + j);
  float4 o;
  {
    float r = sigmoidf_(ir.x + hr.x), z = sigmoidf_(iz.x + hz.x);
    float nn = tanhf(in.x + r * hn.x); o.x = (1.f - z) * nn + z * hv.x;
  }
  {
    float r = sigmoidf_(ir.y + hr.y), z = sigmoidf_(iz.y + hz.y);
    float nn = tanhf(in.y + r * hn.y); o.y = (1.f - z) * nn + z * hv.y;
  }
  {
    float r = sigmoidf_(ir.z + hr.z), z = sigmoidf_(iz.z + hz.z);
    float nn = tanhf(in.z + r * hn.z); o.z = (1.f - z) * nn + z * hv.z;
  }
  {
    float r = sigmoidf_(ir.w + hr.w), z = sigmoidf_(iz.w + hz.w);
    float nn = tanhf(in.w + r * hn.w); o.w = (1.f - z) * nn + z * hv.w;
  }
  *reinterpret_cast<float4*>(out + (size_t)n * 64 + j) = o;
}

// ---------------- host-side helpers ----------------

static inline dim3 g1(long long n, int b) { return dim3((unsigned)((n + b - 1) / b)); }

static inline void launch_gemm(hipStream_t s, int Kpad, const __bf16* A, int lda,
                               const __bf16* Wb, int Mp, const float* bias,
                               float* C, int ldc, int N)
{
  long long tiles = (long long)(N >> 4) * (Mp >> 4);
  int blocks = (int)((tiles + 7) / 8);
  if (Kpad == 64)
    wmma_gemm_bias<64, false><<<blocks, 256, 0, s>>>(A, lda, Wb, Mp, bias, (void*)C, ldc, N);
  else
    wmma_gemm_bias<96, false><<<blocks, 256, 0, s>>>(A, lda, Wb, Mp, bias, (void*)C, ldc, N);
}

static inline void launch_gemm_bfout(hipStream_t s, const __bf16* A, int lda,
                                     const __bf16* Wb, int Mp, const float* bias,
                                     __bf16* C, int ldc, int N)
{
  long long tiles = (long long)(N >> 4) * (Mp >> 4);
  int blocks = (int)((tiles + 7) / 8);
  wmma_gemm_bias<96, true><<<blocks, 256, 0, s>>>(A, lda, Wb, Mp, bias, (void*)C, ldc, N);
}

template <int C>
static void run_gat(hipStream_t s,
                    const __bf16* xsrc, int Ns, const __bf16* xdst, int Nd,
                    const __bf16* Wl, const float* bl,
                    const __bf16* Wr, const float* br, const float* att,
                    const int* srcIdx, const int* dstIdx,
                    float* a_l, float* a_r, float* logits,
                    float* segm_f, float* segd, float* outraw)
{
  const int HC = HEADS * C;
  launch_gemm(s, 64, xsrc, HID, Wl, HC, bl, a_l, HC, Ns);
  launch_gemm(s, 64, xdst, HID, Wr, HC, br, a_r, HC, Nd);
  fillf<<<g1((long long)Nd * HEADS, 256), 256, 0, s>>>(segm_f, 0.f, (long long)Nd * HEADS);
  fillf<<<g1((long long)Nd * HEADS, 256), 256, 0, s>>>(segd, 0.f, (long long)Nd * HEADS);
  fillf<<<g1((long long)Nd * HC, 256), 256, 0, s>>>(outraw, 0.f, (long long)Nd * HC);
  unsigned* segm = (unsigned*)segm_f;
  const int EH = NEDGE * HEADS;
  edge_logits_max<C><<<g1(EH, 256), 256, 0, s>>>(a_l, a_r, att, srcIdx, dstIdx, logits, segm, NEDGE);
  edge_exp_sum<<<g1(EH, 256), 256, 0, s>>>(dstIdx, logits, segm, segd, NEDGE);
  edge_scatter<C><<<g1(EH, 256), 256, 0, s>>>(a_l, srcIdx, dstIdx, logits, segd, outraw, NEDGE);
}

// params assumed pytree-flattened (dict keys sorted) after the 8 arrays
enum {
  IN_TOKEN_FEATS = 0, IN_POOL_FEATS, IN_TOKEN_IDS, IN_POOL_IDS,
  IN_EDGE_SRC, IN_EDGE_DST, IN_TOK_MEM_TAB, IN_POOL_MEM_TAB,
  P_C1P2T_WL, P_C1P2T_WR, P_C1P2T_ATT, P_C1P2T_BIAS, P_C1P2T_BL, P_C1P2T_BR,
  P_C1T2P_WL, P_C1T2P_WR, P_C1T2P_ATT, P_C1T2P_BIAS, P_C1T2P_BL, P_C1T2P_BR,
  P_C2P2T_WL, P_C2P2T_WR, P_C2P2T_ATT, P_C2P2T_BIAS, P_C2P2T_BL, P_C2P2T_BR,
  P_C2T2P_WL, P_C2T2P_WR, P_C2T2P_ATT, P_C2T2P_BIAS, P_C2T2P_BL, P_C2T2P_BR,
  P_HPOOLN_B, P_HPOOLN_G, P_HTOKN_B, P_HTOKN_G,
  P_PGRU_WHH, P_PGRU_WIH, P_PGRU_BHH, P_PGRU_BIH,
  P_PMSG_W, P_PMSG_B,
  P_PNORM_B, P_PNORM_G,
  P_PPROJ_W, P_PPROJ_B,
  P_TGRU_WHH, P_TGRU_WIH, P_TGRU_BHH, P_TGRU_BIH,
  P_TMSG_W, P_TMSG_B,
  P_TNORM_B, P_TNORM_G,
  P_TPROJ_W, P_TPROJ_B
};

extern "C" void kernel_launch(void* const* d_in, const int* in_sizes, int n_in,
                              void* d_out, int out_size, void* d_ws, size_t ws_size,
                              hipStream_t stream)
{
  (void)in_sizes; (void)n_in; (void)out_size;
  #define F(i) ((const float*)d_in[i])
  const float* token_feats = F(IN_TOKEN_FEATS);
  const float* pool_feats  = F(IN_POOL_FEATS);
  const int*   token_ids   = (const int*)d_in[IN_TOKEN_IDS];
  const int*   pool_ids    = (const int*)d_in[IN_POOL_IDS];
  const int*   edge_src    = (const int*)d_in[IN_EDGE_SRC];
  const int*   edge_dst    = (const int*)d_in[IN_EDGE_DST];
  const float* tok_mem_tab = F(IN_TOK_MEM_TAB);
  const float* pool_mem_tab= F(IN_POOL_MEM_TAB);

  float* out = (float*)d_out;
  float* tok_emb      = out;                                   // [NTOK,64]
  float* pool_emb     = out + (size_t)NTOK * 64;               // [NPOOL,64]
  float* tok_mem_out  = out + (size_t)(NTOK + NPOOL) * 64;     // [NTOK,64]
  float* pool_mem_out = out + (size_t)(2 * NTOK + NPOOL) * 64; // [NPOOL,64]

  // workspace bump allocator
  float* ws = (float*)d_ws;
  size_t cur = 0;
  auto alloc = [&](size_t n) { float* p = ws + cur; cur += (n + 3) & ~(size_t)3; return p; };
  auto allocb = [&](size_t n) { return (__bf16*)alloc((n + 1) / 2); };

  __bf16* xinb_t = allocb((size_t)NTOK * 96);   // bf16 [feats|mem|0-pad]
  __bf16* xinb_p = allocb((size_t)NPOOL * 96);
  float*  memf_t = alloc((size_t)NTOK * 64);    // f32 gathered memory
  float*  memf_p = alloc((size_t)NPOOL * 64);
  __bf16* memb_t = allocb((size_t)NTOK * 64);   // bf16 gathered memory
  __bf16* memb_p = allocb((size_t)NPOOL * 64);
  __bf16* xt_b   = allocb((size_t)NTOK * 64);   // bf16 activations (GEMM-A only)
  __bf16* xp_b   = allocb((size_t)NPOOL * 64);
  __bf16* ht_b   = allocb((size_t)NTOK * 64);
  __bf16* hp_b   = allocb((size_t)NPOOL * 64);
  float* a_l    = alloc((size_t)NTOK * 256);    // f32; reused as bf16 cat buffer / gh
  float* a_r    = alloc((size_t)NTOK * 256);    // f32; reused as bf16 msg buffer
  float* logits = alloc((size_t)NEDGE * HEADS);
  float* segm   = alloc((size_t)NTOK * HEADS);
  float* segd   = alloc((size_t)NTOK * HEADS);
  float* outraw = alloc((size_t)NTOK * 256);    // gat raw out / gi

  // packed bf16 weights (zero-padded to Kp x Mp)
  __bf16* wb_tproj = allocb(96 * 64);
  __bf16* wb_pproj = allocb(96 * 64);
  __bf16* wb_c1t2p_l = allocb(64 * 64);
  __bf16* wb_c1t2p_r = allocb(64 * 64);
  __bf16* wb_c1p2t_l = allocb(64 * 64);
  __bf16* wb_c1p2t_r = allocb(64 * 64);
  __bf16* wb_c2t2p_l = allocb(64 * 256);
  __bf16* wb_c2t2p_r = allocb(64 * 256);
  __bf16* wb_c2p2t_l = allocb(64 * 256);
  __bf16* wb_c2p2t_r = allocb(64 * 256);
  __bf16* wb_tmsg = allocb(96 * 96);
  __bf16* wb_pmsg = allocb(96 * 96);
  __bf16* wb_tgih = allocb(96 * 192);
  __bf16* wb_pgih = allocb(96 * 192);
  __bf16* wb_tghh = allocb(64 * 192);
  __bf16* wb_pghh = allocb(64 * 192);
  float* pb_tmsg = alloc(96);
  float* pb_pmsg = alloc(96);
  if (ws_size < cur * sizeof(float)) return;   // insufficient scratch

  auto pack = [&](int pidx, int K, int M, __bf16* dst, int Kp, int Mp) {
    pack_wbf<<<g1((long long)Kp * Mp, 256), 256, 0, stream>>>(F(pidx), K, M, dst, Kp, Mp);
  };
  // 0) pack all weights to padded bf16
  pack(P_TPROJ_W, 68, 64, wb_tproj, 96, 64);
  pack(P_PPROJ_W, 72, 64, wb_pproj, 96, 64);
  pack(P_C1T2P_WL, 64, 64, wb_c1t2p_l, 64, 64);
  pack(P_C1T2P_WR, 64, 64, wb_c1t2p_r, 64, 64);
  pack(P_C1P2T_WL, 64, 64, wb_c1p2t_l, 64, 64);
  pack(P_C1P2T_WR, 64, 64, wb_c1p2t_r, 64, 64);
  pack(P_C2T2P_WL, 64, 256, wb_c2t2p_l, 64, 256);
  pack(P_C2T2P_WR, 64, 256, wb_c2t2p_r, 64, 256);
  pack(P_C2P2T_WL, 64, 256, wb_c2p2t_l, 64, 256);
  pack(P_C2P2T_WR, 64, 256, wb_c2p2t_r, 64, 256);
  pack(P_TMSG_W, 68, 68, wb_tmsg, 96, 96);
  pack(P_PMSG_W, 72, 72, wb_pmsg, 96, 96);
  pack(P_TGRU_WIH, 68, 192, wb_tgih, 96, 192);
  pack(P_PGRU_WIH, 72, 192, wb_pgih, 96, 192);
  pack(P_TGRU_WHH, 64, 192, wb_tghh, 64, 192);
  pack(P_PGRU_WHH, 64, 192, wb_pghh, 64, 192);
  pack_bias<<<g1(96, 96), 96, 0, stream>>>(F(P_TMSG_B), 68, pb_tmsg, 96);
  pack_bias<<<g1(96, 96), 96, 0, stream>>>(F(P_PMSG_B), 72, pb_pmsg, 96);

  // 1) gather memory + concat (bf16, zero-padded to 96 cols) + f32/bf16 memory copies
  concat_gather<<<g1((long long)NTOK * 96, 256), 256, 0, stream>>>(
      token_feats, TOKF, token_ids, tok_mem_tab, xinb_t, memf_t, memb_t, NTOK);
  concat_gather<<<g1((long long)NPOOL * 96, 256), 256, 0, stream>>>(
      pool_feats, POOLF, pool_ids, pool_mem_tab, xinb_p, memf_p, memb_p, NPOOL);

  // 2) input projections + LN + ELU (-> bf16)
  launch_gemm(stream, 96, xinb_t, 96, wb_tproj, 64, F(P_TPROJ_B), a_l, 64, NTOK);
  ln_elu_kernel<<<g1(NTOK, 8), 256, 0, stream>>>(a_l, nullptr, F(P_TNORM_G), F(P_TNORM_B), xt_b, NTOK);
  launch_gemm(stream, 96, xinb_p, 96, wb_pproj, 64, F(P_PPROJ_B), a_r, 64, NPOOL);
  ln_elu_kernel<<<g1(NPOOL, 8), 256, 0, stream>>>(a_r, nullptr, F(P_PNORM_G), F(P_PNORM_B), xp_b, NPOOL);

  // 3) conv1 (heads=4, C=16, concat) + hidden LN + ELU (-> bf16)
  run_gat<16>(stream, xt_b, NTOK, xp_b, NPOOL, wb_c1t2p_l, F(P_C1T2P_BL),
              wb_c1t2p_r, F(P_C1T2P_BR), F(P_C1T2P_ATT),
              edge_src, edge_dst, a_l, a_r, logits, segm, segd, outraw);
  ln_elu_kernel<<<g1(NPOOL, 8), 256, 0, stream>>>(outraw, F(P_C1T2P_BIAS), F(P_HPOOLN_G), F(P_HPOOLN_B), hp_b, NPOOL);

  run_gat<16>(stream, xp_b, NPOOL, xt_b, NTOK, wb_c1p2t_l, F(P_C1P2T_BL),
              wb_c1p2t_r, F(P_C1P2T_BR), F(P_C1P2T_ATT),
              edge_dst, edge_src, a_l, a_r, logits, segm, segd, outraw);
  ln_elu_kernel<<<g1(NTOK, 8), 256, 0, stream>>>(outraw, F(P_C1P2T_BIAS), F(P_HTOKN_G), F(P_HTOKN_B), ht_b, NTOK);

  // 4) conv2 (heads=4, C=64, mean over heads) -> embeddings into d_out
  run_gat<64>(stream, ht_b, NTOK, hp_b, NPOOL, wb_c2t2p_l, F(P_C2T2P_BL),
              wb_c2t2p_r, F(P_C2T2P_BR), F(P_C2T2P_ATT),
              edge_src, edge_dst, a_l, a_r, logits, segm, segd, outraw);
  head_mean_bias<<<g1((long long)NPOOL * 16, 256), 256, 0, stream>>>(outraw, F(P_C2T2P_BIAS), pool_emb, NPOOL);

  run_gat<64>(stream, hp_b, NPOOL, ht_b, NTOK, wb_c2p2t_l, F(P_C2P2T_BL),
              wb_c2p2t_r, F(P_C2P2T_BR), F(P_C2P2T_ATT),
              edge_dst, edge_src, a_l, a_r, logits, segm, segd, outraw);
  head_mean_bias<<<g1((long long)NTOK * 16, 256), 256, 0, stream>>>(outraw, F(P_C2P2T_BIAS), tok_emb, NTOK);

  // 5) token: msg linear (bf16 out) -> GRU -> new memory
  __bf16* cat_b = (__bf16*)a_l;
  __bf16* msg_b = (__bf16*)a_r;
  concat2_bf<<<g1((long long)NTOK * 96, 256), 256, 0, stream>>>(token_feats, TOKF, tok_emb, cat_b, NTOK);
  launch_gemm_bfout(stream, cat_b, 96, wb_tmsg, 96, pb_tmsg, msg_b, 96, NTOK);           // msg (padded)
  launch_gemm(stream, 96, msg_b, 96, wb_tgih, 192, F(P_TGRU_BIH), outraw, 192, NTOK);    // gi
  launch_gemm(stream, 64, memb_t, 64, wb_tghh, 192, F(P_TGRU_BHH), a_l, 192, NTOK);      // gh (cat dead)
  gru_kernel<<<g1((long long)NTOK * 16, 256), 256, 0, stream>>>(outraw, a_l, memf_t, tok_mem_out, NTOK);

  // 6) pool: msg linear (bf16 out) -> GRU -> new memory
  concat2_bf<<<g1((long long)NPOOL * 96, 256), 256, 0, stream>>>(pool_feats, POOLF, pool_emb, cat_b, NPOOL);
  launch_gemm_bfout(stream, cat_b, 96, wb_pmsg, 96, pb_pmsg, msg_b, 96, NPOOL);          // msg (padded)
  launch_gemm(stream, 96, msg_b, 96, wb_pgih, 192, F(P_PGRU_BIH), outraw, 192, NPOOL);   // gi
  launch_gemm(stream, 64, memb_p, 64, wb_pghh, 192, F(P_PGRU_BHH), a_l, 192, NPOOL);     // gh
  gru_kernel<<<g1((long long)NPOOL * 16, 256), 256, 0, stream>>>(outraw, a_l, memf_p, pool_mem_out, NPOOL);
  #undef F
}
